// VoronoiFNO2dSparseSWE_34634616275403
// MI455X (gfx1250) — matrix operand order repaired and checked by reading, not compiled, not run on hardware
//
#include <hip/hip_runtime.h>
#include <math.h>

typedef __attribute__((ext_vector_type(16))) _Float16 v16h;
typedef __attribute__((ext_vector_type(8)))  _Float16 v8h;
typedef __attribute__((ext_vector_type(8)))  float    v8f;

#define NXg 128
#define NYg 128
#define NXY (NXg*NYg)
#define Bn 64
#define Wd 64
#define M1 12
#define M2 12
#define NMODE 144     // 12*12 per corner
#define KNB 64
#define NTc 50

__device__ __forceinline__ float gelu_f(float x) {
  return 0.5f * x * (1.0f + erff(x * 0.70710678118654752f));
}

// A-fragment: 16-bit A 16x32. Lanes 0-15: row M=lane, K = kb+{0..15}.
// Lanes 16-31: row M=lane-16, K = kb+{16..31}.  ld fixed at 64.
__device__ __forceinline__ v16h ld_a(const _Float16* base, int row0, int kb, int lane) {
  int row = row0 + (lane & 15);
  int g   = lane >> 4;
  const _Float16* p = base + (size_t)row * 64 + kb + g * 16;
  union { v16h v; v8h h[2]; } u;
  u.h[0] = *(const v8h*)p;
  u.h[1] = *(const v8h*)(p + 8);
  return u.v;
}

// B-fragment: 16-bit B 32x16 (KxN). Lane: n = lane&15, K = kb + (lane>>4)*16 + 0..15
// Weight stored transposed: Wt[n][c], ld = 64 -> 16 consecutive halves.
__device__ __forceinline__ v16h ld_b(const _Float16* base, int col0, int kb, int lane) {
  int n = col0 + (lane & 15);
  int k = kb + ((lane >> 4) << 4);
  return *(const v16h*)(base + (size_t)n * 64 + k);
}

// ---------------- scatter: build num/den grids ----------------
__global__ void k_scatter(const float* xyt_q, const float* obs_coords, const float* obs_vals,
                          const int* nb_idx, const int* sy, const int* sx,
                          float* num, float* den) {
  int b = blockIdx.x, k = threadIdx.x;
  int idx = nb_idx[b * KNB + k];
  int sid = idx / NTc;
  float w = expf(-0.1f * fabsf(obs_coords[idx * 3 + 2] - xyt_q[b * 3 + 2]));
  float val = obs_vals[idx] * w;
  int lin = sy[sid] * NYg + sx[sid];
  atomicAdd(&num[(size_t)b * NXY + lin], val);
  atomicAdd(&den[(size_t)b * NXY + lin], w);
}

// ---------------- fc0 lift 3->64, store f16 NHWC ----------------
__global__ void k_fc0(const float* num, const float* den, const float* xg, const float* yg,
                      const float* fc0w, const float* fc0b, _Float16* act) {
  int p = blockIdx.x * blockDim.x + threadIdx.x;
  if (p >= Bn * NXY) return;
  int xy = p & (NXY - 1);
  float d = den[p];
  float g = (d > 0.f) ? num[p] / fmaxf(d, 1e-6f) : 0.f;
  float a = xg[xy], c = yg[xy];
  for (int w = 0; w < Wd; ++w) {
    float v = g * fc0w[w] + a * fc0w[64 + w] + c * fc0w[128 + w] + fc0b[w];
    act[(size_t)p * 64 + w] = (_Float16)v;
  }
}

// ---------------- weight prep: spec weights -> f16 [l][corner][m][o][c] ----------------
__global__ void k_prep_spec(const float* w1r, const float* w1i, const float* w2r, const float* w2i,
                            _Float16* whr, _Float16* whi) {
  size_t i = (size_t)blockIdx.x * blockDim.x + threadIdx.x;
  if (i >= (size_t)4 * 2 * NMODE * 64 * 64) return;
  int c = (int)(i & 63);
  int o = (int)((i >> 6) & 63);
  int m = (int)((i >> 12) % NMODE);
  int lc = (int)((i >> 12) / NMODE);
  int corner = lc & 1, l = lc >> 1;
  size_t src = (((size_t)l * 64 + c) * 64 + o) * NMODE + m;
  whr[i] = (_Float16)(corner ? w2r[src] : w1r[src]);
  whi[i] = (_Float16)(corner ? w2i[src] : w1i[src]);
}

__global__ void k_prep_misc(const float* pw_w, const float* fc1_w, _Float16* pwh, _Float16* fc1t) {
  int i = blockIdx.x * blockDim.x + threadIdx.x;
  if (i < 4 * 64 * 64) {
    pwh[i] = (_Float16)pw_w[i];           // already [l][o][c]
  } else {
    int j = i - 4 * 64 * 64;
    if (j < 128 * 64) {
      int f = j >> 6, c = j & 63;
      fc1t[j] = (_Float16)fc1_w[c * 128 + f];   // transpose (w,f)->(f,w)
    }
  }
}

// ---------------- forward y-DFT (NY=128 -> 12 modes), fold 1/NY ----------------
__global__ void __launch_bounds__(256) k_yfwd(const _Float16* act, float* yfr, float* yfi) {
  __shared__ float srow[NYg * Wd];
  __shared__ float sct[M2 * NYg];
  __shared__ float sst[M2 * NYg];
  int bx = blockIdx.x;                 // b*128 + x
  const _Float16* src = act + (size_t)bx * NYg * Wd;
  for (int i = threadIdx.x; i < NYg * Wd; i += 256) srow[i] = (float)src[i];
  for (int i = threadIdx.x; i < M2 * NYg; i += 256) {
    int ky = i >> 7, y = i & 127;
    float s, cc;
    __sincosf(6.283185307179586f * (float)(ky * y) * (1.0f / 128.0f), &s, &cc);
    sct[i] = cc * (1.0f / 128.0f);
    sst[i] = s  * (1.0f / 128.0f);
  }
  __syncthreads();
  for (int j = 0; j < 3; ++j) {
    int pair = threadIdx.x + 256 * j;  // 768 = 64 ch * 12 modes
    int ch = pair / M2, ky = pair % M2;
    float ar = 0.f, ai = 0.f;
    for (int y = 0; y < NYg; ++y) {
      float v = srow[y * Wd + ch];
      ar += v * sct[ky * 128 + y];
      ai -= v * sst[ky * 128 + y];
    }
    size_t o = ((size_t)bx * Wd + ch) * M2 + ky;
    yfr[o] = ar; yfi[o] = ai;
  }
}

// ---------------- forward x-DFT (128 -> 24 corner modes), fold 1/NX, write f16 ----------------
__global__ void __launch_bounds__(256) k_xfwd(const float* yfr, const float* yfi,
                                              _Float16* xfr, _Float16* xfi) {
  __shared__ float cs[NXg], sn[NXg];
  int cm = blockIdx.x;                 // 0..287 = corner*144 + m
  int corner = cm / NMODE, m = cm % NMODE;
  int kx = (corner ? 116 : 0) + m / M2;
  int kxe = (kx >= 64) ? kx - 128 : kx;
  int ky = m % M2;
  for (int x = threadIdx.x; x < NXg; x += 256) {
    float s, cc;
    __sincosf(6.283185307179586f * (float)(kxe * x) * (1.0f / 128.0f), &s, &cc);
    cs[x] = cc * (1.0f / 128.0f);
    sn[x] = s  * (1.0f / 128.0f);
  }
  __syncthreads();
  for (int j = 0; j < 2; ++j) {        // blockIdx.y picks 512 of the 4096 (b,c) pairs
    int bc = threadIdx.x + 256 * (blockIdx.y * 2 + j);
    int b = bc >> 6, c = bc & 63;
    float xr = 0.f, xi = 0.f;
    const float* pr = yfr + ((size_t)b * NXg * Wd + c) * M2 + ky;
    const float* pi = yfi + ((size_t)b * NXg * Wd + c) * M2 + ky;
    for (int x = 0; x < NXg; ++x) {
      float yr = pr[(size_t)x * Wd * M2];
      float yi = pi[(size_t)x * Wd * M2];
      xr += yr * cs[x] + yi * sn[x];
      xi += yi * cs[x] - yr * sn[x];
    }
    xfr[(size_t)cm * 4096 + bc] = (_Float16)xr;
    xfi[(size_t)cm * 4096 + bc] = (_Float16)xi;
  }
}

// ---------------- per-mode complex 64x64 channel mix (WMMA, A-frags reused over 4 N-tiles) ----
__global__ void __launch_bounds__(256) k_mix(const _Float16* xfr, const _Float16* xfi,
                                             const _Float16* whr, const _Float16* whi,
                                             _Float16* ofr, _Float16* ofi) {
  int wid  = blockIdx.x * 8 + (threadIdx.x >> 5);
  int lane = threadIdx.x & 31;
  int cm = wid >> 2;                   // corner*144 + mode
  int m0 = (wid & 3) * 16;             // M tile (batch rows)
  const _Float16* ab_r = xfr + (size_t)cm * 4096;
  const _Float16* ab_i = xfi + (size_t)cm * 4096;
  const _Float16* wb_r = whr + (size_t)cm * 4096;
  const _Float16* wb_i = whi + (size_t)cm * 4096;
  v16h ar0 = ld_a(ab_r, m0, 0, lane);
  v16h ar1 = ld_a(ab_r, m0, 32, lane);
  v16h ai0 = ld_a(ab_i, m0, 0, lane);
  v16h ai1 = ld_a(ab_i, m0, 32, lane);
  int mrow = m0 + 8 * (lane >> 4);
  for (int nt = 0; nt < 4; ++nt) {
    int n0 = nt * 16;
    v16h br0 = ld_b(wb_r, n0, 0, lane);
    v16h br1 = ld_b(wb_r, n0, 32, lane);
    v16h bi0 = ld_b(wb_i, n0, 0, lane);
    v16h bi1 = ld_b(wb_i, n0, 32, lane);
    v8f crr = {}, cii = {}, cri = {}, cir = {};
    crr = __builtin_amdgcn_wmma_f32_16x16x32_f16(false, ar0, false, br0, (short)0, crr, false, false);
    crr = __builtin_amdgcn_wmma_f32_16x16x32_f16(false, ar1, false, br1, (short)0, crr, false, false);
    cii = __builtin_amdgcn_wmma_f32_16x16x32_f16(false, ai0, false, bi0, (short)0, cii, false, false);
    cii = __builtin_amdgcn_wmma_f32_16x16x32_f16(false, ai1, false, bi1, (short)0, cii, false, false);
    cri = __builtin_amdgcn_wmma_f32_16x16x32_f16(false, ar0, false, bi0, (short)0, cri, false, false);
    cri = __builtin_amdgcn_wmma_f32_16x16x32_f16(false, ar1, false, bi1, (short)0, cri, false, false);
    cir = __builtin_amdgcn_wmma_f32_16x16x32_f16(false, ai0, false, br0, (short)0, cir, false, false);
    cir = __builtin_amdgcn_wmma_f32_16x16x32_f16(false, ai1, false, br1, (short)0, cir, false, false);
    int n = n0 + (lane & 15);
    for (int v = 0; v < 8; ++v) {
      size_t o = (size_t)cm * 4096 + (size_t)(mrow + v) * 64 + n;
      ofr[o] = (_Float16)(crr[v] - cii[v]);
      ofi[o] = (_Float16)(cri[v] + cir[v]);
    }
  }
}

// ---------------- inverse x-DFT: 24 modes -> 128 x, complex (shared basis per block) --------
__global__ void __launch_bounds__(256) k_xinv(const _Float16* ofr, const _Float16* ofi,
                                              float* gr, float* gi) {
  __shared__ float cs[24], sn[24];
  int i0 = blockIdx.x * 256;
  int x = (i0 / (M2 * 64)) & 127;      // constant over block: 768 = 3*256
  if (threadIdx.x < 24) {
    int kxe = (threadIdx.x < 12) ? threadIdx.x : (int)threadIdx.x - 24;
    float s, cc;
    __sincosf(6.283185307179586f * (float)(kxe * x) * (1.0f / 128.0f), &s, &cc);
    cs[threadIdx.x] = cc; sn[threadIdx.x] = s;
  }
  __syncthreads();
  int i = i0 + threadIdx.x;
  if (i >= Bn * NXg * Wd * M2) return;
  int ky = i % M2;
  int o  = (i / M2) & 63;
  int b  = i / (M2 * 64 * 128);
  float sr = 0.f, si = 0.f;
  for (int j = 0; j < 24; ++j) {
    int cm = ((j < 12) ? 0 : NMODE) + (j % 12) * M2 + ky;
    float pr = (float)ofr[(size_t)cm * 4096 + (size_t)b * 64 + o];
    float pi = (float)ofi[(size_t)cm * 4096 + (size_t)b * 64 + o];
    sr += pr * cs[j] - pi * sn[j];
    si += pr * sn[j] + pi * cs[j];
  }
  gr[i] = sr; gi[i] = si;
}

// ---------------- pointwise conv (WMMA, 16x64 per wave) + inverse-y spec + bias + GELU ------
__global__ void __launch_bounds__(256) k_pw_spec(const _Float16* actIn, const _Float16* pwh,
                                                 const float* pwb, const float* gr, const float* gi,
                                                 _Float16* actOut, int do_gelu) {
  __shared__ float sct[M2 * NYg];
  __shared__ float sst[M2 * NYg];
  for (int i = threadIdx.x; i < M2 * NYg; i += 256) {
    int ky = i >> 7, y = i & 127;
    float s, cc;
    __sincosf(6.283185307179586f * (float)(ky * y) * (1.0f / 128.0f), &s, &cc);
    float alpha = (ky == 0) ? 1.0f : 2.0f;
    sct[i] = alpha * cc; sst[i] = alpha * s;
  }
  __syncthreads();
  int wid  = blockIdx.x * 8 + (threadIdx.x >> 5);   // one wave per 16-row slab
  int lane = threadIdx.x & 31;
  int m0 = wid * 16;
  __builtin_prefetch(actIn + (size_t)(m0 + 128) * 64, 0, 1);   // stream ahead
  v16h a0 = ld_a(actIn, m0, 0, lane);
  v16h a1 = ld_a(actIn, m0, 32, lane);
  int mrow = m0 + 8 * (lane >> 4);
  int b  = mrow >> 14;
  int xy = mrow & (NXY - 1);
  int xx = xy >> 7, yy0 = xy & 127;    // 8 consecutive yy share (b, xx)
  for (int nt = 0; nt < 4; ++nt) {
    v8f acc = {};
    v16h b0 = ld_b(pwh, nt * 16, 0, lane);
    v16h b1 = ld_b(pwh, nt * 16, 32, lane);
    acc = __builtin_amdgcn_wmma_f32_16x16x32_f16(false, a0, false, b0, (short)0, acc, false, false);
    acc = __builtin_amdgcn_wmma_f32_16x16x32_f16(false, a1, false, b1, (short)0, acc, false, false);
    int o = nt * 16 + (lane & 15);
    float bias = pwb[o];
    const float* pgr = gr + (((size_t)(b * NXg + xx)) * 64 + o) * M2;
    const float* pgi = gi + (((size_t)(b * NXg + xx)) * 64 + o) * M2;
    float grv[M2], giv[M2];
    for (int ky = 0; ky < M2; ++ky) { grv[ky] = pgr[ky]; giv[ky] = pgi[ky]; }
    for (int v = 0; v < 8; ++v) {
      int yy = yy0 + v;
      float spec = 0.f;
      for (int ky = 0; ky < M2; ++ky)
        spec += grv[ky] * sct[ky * 128 + yy] - giv[ky] * sst[ky * 128 + yy];
      float val = acc[v] + spec + bias;
      if (do_gelu) val = gelu_f(val);
      actOut[(size_t)(mrow + v) * 64 + o] = (_Float16)val;
    }
  }
}

// ---------------- fused fc1 (WMMA) + GELU + fc2 (shuffle reduce) ----------------
__global__ void __launch_bounds__(256) k_head(const _Float16* act, const _Float16* fc1t,
                                              const float* fc1b, const float* fc2w, const float* fc2b,
                                              float* field) {
  int wid  = blockIdx.x * 8 + (threadIdx.x >> 5);
  int lane = threadIdx.x & 31;
  int m0 = wid * 16;
  __builtin_prefetch(act + (size_t)(m0 + 128) * 64, 0, 1);
  v16h a0 = ld_a(act, m0, 0, lane);
  v16h a1 = ld_a(act, m0, 32, lane);
  float part[8][3] = {};
  for (int t = 0; t < 8; ++t) {
    v8f acc = {};
    v16h b0 = ld_b(fc1t, t * 16, 0, lane);
    v16h b1 = ld_b(fc1t, t * 16, 32, lane);
    acc = __builtin_amdgcn_wmma_f32_16x16x32_f16(false, a0, false, b0, (short)0, acc, false, false);
    acc = __builtin_amdgcn_wmma_f32_16x16x32_f16(false, a1, false, b1, (short)0, acc, false, false);
    int col = t * 16 + (lane & 15);
    float fb = fc1b[col];
    float w0 = fc2w[col * 3 + 0], w1 = fc2w[col * 3 + 1], w2 = fc2w[col * 3 + 2];
    for (int v = 0; v < 8; ++v) {
      float h = gelu_f(acc[v] + fb);
      part[v][0] += h * w0; part[v][1] += h * w1; part[v][2] += h * w2;
    }
  }
  for (int m = 1; m < 16; m <<= 1)
    for (int v = 0; v < 8; ++v)
      for (int o = 0; o < 3; ++o)
        part[v][o] += __shfl_xor(part[v][o], m, 32);
  if ((lane & 15) == 0) {
    int base = m0 + 8 * (lane >> 4);
    for (int v = 0; v < 8; ++v)
      for (int o = 0; o < 3; ++o)
        field[(size_t)(base + v) * 3 + o] = part[v][o] + fc2b[o];
  }
}

// ---------------- bilinear sample at query points ----------------
__global__ void k_sample(const float* field, const float* xyt_q, const int* Lx, const int* Ly,
                         float* out) {
  int b = threadIdx.x;
  if (b >= Bn) return;
  float lx = fmaxf((float)Lx[0], 1e-6f);
  float ly = fmaxf((float)Ly[0], 1e-6f);
  float x01 = fminf(fmaxf(xyt_q[b * 3 + 0] / lx, 0.f), 1.f);
  float y01 = fminf(fmaxf(xyt_q[b * 3 + 1] / ly, 0.f), 1.f);
  float gx = x01 * (NXg - 1), gy = y01 * (NYg - 1);
  int x0 = (int)floorf(gx), y0 = (int)floorf(gy);
  int x1 = min(x0 + 1, NXg - 1), y1 = min(y0 + 1, NYg - 1);
  float wx = gx - (float)x0, wy = gy - (float)y0;
  for (int o = 0; o < 3; ++o) {
    float f00 = field[(((size_t)b * NXg + x0) * NYg + y0) * 3 + o];
    float f10 = field[(((size_t)b * NXg + x1) * NYg + y0) * 3 + o];
    float f01 = field[(((size_t)b * NXg + x0) * NYg + y1) * 3 + o];
    float f11 = field[(((size_t)b * NXg + x1) * NYg + y1) * 3 + o];
    float top = f00 * (1.f - wx) + f10 * wx;
    float bot = f01 * (1.f - wx) + f11 * wx;
    out[b * 3 + o] = top * (1.f - wy) + bot * wy;
  }
}

extern "C" void kernel_launch(void* const* d_in, const int* in_sizes, int n_in,
                              void* d_out, int out_size, void* d_ws, size_t ws_size,
                              hipStream_t stream) {
  (void)in_sizes; (void)n_in; (void)out_size;
  const float* xyt_q      = (const float*)d_in[0];
  const float* obs_coords = (const float*)d_in[1];
  const float* obs_vals   = (const float*)d_in[2];
  const float* x_grid     = (const float*)d_in[3];
  const float* y_grid     = (const float*)d_in[4];
  const float* fc0_w      = (const float*)d_in[5];
  const float* fc0_b      = (const float*)d_in[6];
  const float* w1r        = (const float*)d_in[7];
  const float* w1i        = (const float*)d_in[8];
  const float* w2r        = (const float*)d_in[9];
  const float* w2i        = (const float*)d_in[10];
  const float* pw_w       = (const float*)d_in[11];
  const float* pw_b       = (const float*)d_in[12];
  const float* fc1_w      = (const float*)d_in[13];
  const float* fc1_b      = (const float*)d_in[14];
  const float* fc2_w      = (const float*)d_in[15];
  const float* fc2_b      = (const float*)d_in[16];
  const int*   nb_idx     = (const int*)d_in[17];
  const int*   sy         = (const int*)d_in[18];
  const int*   sx         = (const int*)d_in[19];
  const int*   Lx         = (const int*)d_in[20];
  const int*   Ly         = (const int*)d_in[21];

  char* wsb = (char*)d_ws;
  size_t off = 0;
  auto take = [&](size_t bytes) -> char* {
    char* p = wsb + off;
    off = (off + bytes + 255) & ~(size_t)255;
    return p;
  };
  float*     num  = (float*)take((size_t)Bn * NXY * 4);
  float*     den  = (float*)take((size_t)Bn * NXY * 4);
  _Float16*  actA = (_Float16*)take((size_t)Bn * NXY * Wd * 2);
  _Float16*  actB = (_Float16*)take((size_t)Bn * NXY * Wd * 2);
  float*     yfr  = (float*)take((size_t)Bn * NXg * Wd * M2 * 4);   // reused as Gr
  float*     yfi  = (float*)take((size_t)Bn * NXg * Wd * M2 * 4);   // reused as Gi
  _Float16*  xfr  = (_Float16*)take((size_t)2 * NMODE * 4096 * 2);
  _Float16*  xfi  = (_Float16*)take((size_t)2 * NMODE * 4096 * 2);
  _Float16*  ofr  = (_Float16*)take((size_t)2 * NMODE * 4096 * 2);
  _Float16*  ofi  = (_Float16*)take((size_t)2 * NMODE * 4096 * 2);
  _Float16*  whr  = (_Float16*)take((size_t)4 * 2 * NMODE * 4096 * 2);
  _Float16*  whi  = (_Float16*)take((size_t)4 * 2 * NMODE * 4096 * 2);
  _Float16*  pwh  = (_Float16*)take((size_t)4 * 64 * 64 * 2);
  _Float16*  fc1t = (_Float16*)take((size_t)128 * 64 * 2);
  float*     field = (float*)take((size_t)Bn * NXY * 3 * 4);
  if (off > ws_size) return;

  hipMemsetAsync(num, 0, (size_t)Bn * NXY * 4, stream);
  hipMemsetAsync(den, 0, (size_t)Bn * NXY * 4, stream);

  k_scatter<<<Bn, KNB, 0, stream>>>(xyt_q, obs_coords, obs_vals, nb_idx, sy, sx, num, den);
  k_fc0<<<(Bn * NXY) / 256, 256, 0, stream>>>(num, den, x_grid, y_grid, fc0_w, fc0_b, actA);

  {
    size_t tot = (size_t)4 * 2 * NMODE * 4096;
    k_prep_spec<<<(unsigned)((tot + 255) / 256), 256, 0, stream>>>(w1r, w1i, w2r, w2i, whr, whi);
    k_prep_misc<<<(4 * 64 * 64 + 128 * 64 + 255) / 256, 256, 0, stream>>>(pw_w, fc1_w, pwh, fc1t);
  }

  _Float16* cin = actA;
  _Float16* cout = actB;
  for (int l = 0; l < 4; ++l) {
    k_yfwd<<<Bn * NXg, 256, 0, stream>>>(cin, yfr, yfi);
    k_xfwd<<<dim3(2 * NMODE, 8), 256, 0, stream>>>(yfr, yfi, xfr, xfi);
    k_mix<<<(2 * NMODE * 4) / 8, 256, 0, stream>>>(
        xfr, xfi, whr + (size_t)l * 2 * NMODE * 4096, whi + (size_t)l * 2 * NMODE * 4096, ofr, ofi);
    k_xinv<<<(Bn * NXg * Wd * M2) / 256, 256, 0, stream>>>(ofr, ofi, yfr, yfi);
    k_pw_spec<<<(Bn * NXY / 16) / 8, 256, 0, stream>>>(
        cin, pwh + (size_t)l * 4096, pw_b + l * 64, yfr, yfi, cout, (l < 3) ? 1 : 0);
    _Float16* t = cin; cin = cout; cout = t;
  }

  k_head<<<(Bn * NXY / 16) / 8, 256, 0, stream>>>(cin, fc1t, fc1_b, fc2_w, fc2_b, field);
  k_sample<<<1, 64, 0, stream>>>(field, xyt_q, Lx, Ly, (float*)d_out);
}
